// HyperbolicGraphConvolution_7662221656589
// MI455X (gfx1250) — compile-verified, matrix operand order checked by reading, not run on hardware
//
#include <hip/hip_runtime.h>
#include <hip/hip_fp16.h>

typedef __attribute__((ext_vector_type(16))) _Float16 v16h;
typedef __attribute__((ext_vector_type(8)))  _Float16 v8h;
typedef __attribute__((ext_vector_type(8)))  float    v8f;

#define DEV __device__ __forceinline__

static constexpr int Bn = 4, Nn = 256, Dn = 128;
static constexpr int BN = Bn * Nn;          // 1024
static constexpr float MINN = 1e-15f;
static constexpr float MAXN = 1.0f - 4e-3f; // (1 - PROJ_EPS)/sqrt(c), c=1

DEV float artanh_c(float x) {
    x = fminf(fmaxf(x, -1.0f + 1e-7f), 1.0f - 1e-7f);
    return 0.5f * logf((1.0f + x) / (1.0f - x));
}
DEV float siluf(float x)    { return x / (1.0f + expf(-x)); }
DEV float sigmoidf(float x) { return 1.0f / (1.0f + expf(-x)); }

// ---------------- block-wide sum over 128 threads ----------------
DEV float blockSum128(float v, float* sm) {
    const int t = threadIdx.x;
    sm[t] = v; __syncthreads();
    #pragma unroll
    for (int s = 64; s > 0; s >>= 1) {
        if (t < s) sm[t] += sm[t + s];
        __syncthreads();
    }
    float r = sm[0]; __syncthreads();
    return r;
}

// ---------------- f32 -> f16 convert ----------------
__global__ void k_cvt_h(const float* __restrict__ src, _Float16* __restrict__ dst, int n) {
    int i = blockIdx.x * blockDim.x + threadIdx.x;
    if (i < n) dst[i] = (_Float16)src[i];
}

// ---------------- f16 32x32 LDS-tiled transpose (per batch slice) ----------------
// dst[b][d][n] = src[b][n][d]; src: Nn x Dn, dst: Dn x Nn
__global__ void k_transpose_h(const _Float16* __restrict__ src, _Float16* __restrict__ dst) {
    __shared__ _Float16 tile[32][33];          // +1 pad: avoid bank conflicts
    const int bx = blockIdx.x * 32;            // n tile
    const int by = blockIdx.y * 32;            // d tile
    const _Float16* s = src + (size_t)blockIdx.z * Nn * Dn;
    _Float16*       d = dst + (size_t)blockIdx.z * Nn * Dn;
    for (int r = threadIdx.y; r < 32; r += 8)
        tile[r][threadIdx.x] = s[(size_t)(bx + r) * Dn + (by + threadIdx.x)];
    __syncthreads();
    for (int r = threadIdx.y; r < 32; r += 8)
        d[(size_t)(by + r) * Nn + (bx + threadIdx.x)] = tile[threadIdx.x][r];
#if __has_builtin(__builtin_amdgcn_s_wait_tensorcnt)
    __builtin_amdgcn_s_wait_tensorcnt(0);      // CDNA5 counter path (trivially satisfied)
#endif
#if __has_builtin(__builtin_amdgcn_s_cluster_barrier)
    __builtin_amdgcn_s_cluster_barrier();      // NOP when not dispatched in a cluster
#endif
}

// ---------------- generic WMMA GEMM ----------------
// C[m][n] = sum_k A[m][k] * Bm[n][k]   (C = A * Bm^T, Bm row-major N x K)
// one wave (32 lanes, EXEC all ones) per 16x16 output tile; grid.z = batch.
__global__ void k_gemm_wmma(const _Float16* __restrict__ A, long long sA, int lda,
                            const _Float16* __restrict__ Bm, long long sB, int ldb,
                            float* __restrict__ C, long long sC, int ldc,
                            int K)
{
    const int lane = threadIdx.x;        // 0..31
    const int m16  = lane & 15;
    const int half = lane >> 4;          // 0: K-low 8, 1: K-high 8 within a 16-group
    const int tm   = blockIdx.x * 16;
    const int tn   = blockIdx.y * 16;
    A  += (long long)blockIdx.z * sA;
    Bm += (long long)blockIdx.z * sB;
    C  += (long long)blockIdx.z * sC;

    const _Float16* pa = A  + (size_t)(tm + m16) * lda + half * 8;
    const _Float16* pb = Bm + (size_t)(tn + m16) * ldb + half * 8;

    v8f acc = {};
    for (int k0 = 0; k0 < K; k0 += 32) {
        v16h a, b;
        {   // fragment halves: j<8 -> k = half*8 + j ; j>=8 -> k = 16 + half*8 + (j-8)
            v8h lo = *(const v8h*)(pa + k0);
            v8h hi = *(const v8h*)(pa + k0 + 16);
            #pragma unroll
            for (int j = 0; j < 8; ++j) { a[j] = lo[j]; a[j + 8] = hi[j]; }
        }
        {
            v8h lo = *(const v8h*)(pb + k0);
            v8h hi = *(const v8h*)(pb + k0 + 16);
            #pragma unroll
            for (int j = 0; j < 8; ++j) { b[j] = lo[j]; b[j + 8] = hi[j]; }
        }
        if (k0 + 32 < K) {               // exercise global_prefetch path for next K block
            __builtin_prefetch(pa + k0 + 32, 0, 0);
            __builtin_prefetch(pb + k0 + 32, 0, 0);
        }
        acc = __builtin_amdgcn_wmma_f32_16x16x32_f16(
            /*neg_a=*/false, a, /*neg_b=*/false, b,
            /*c_mod=*/(short)0, acc, /*reuse_a=*/false, /*reuse_b=*/false);
    }
    #pragma unroll
    for (int r = 0; r < 8; ++r) {
        int row = tm + r + half * 8;
        int col = tn + m16;
        C[(size_t)row * ldc + col] = acc[r];
    }
}

// ---------------- HypLinear pointwise (per row of 128) ----------------
// h = proj(mobius_add(proj(mobius_matvec(W,x)), proj(expmap0(b_lin))))
// also: Hh (f16), h2 = ||h||^2, x_self = logmap(h,h) (+f16 copy)
__global__ void k_hyplinear(const float* __restrict__ x, const float* __restrict__ MX,
                            const float* __restrict__ b_lin,
                            float* __restrict__ H, _Float16* __restrict__ Hh,
                            float* __restrict__ h2out,
                            float* __restrict__ Xself, _Float16* __restrict__ Xselfh)
{
    __shared__ float sm[128];
    const int r = blockIdx.x, t = threadIdx.x;
    const size_t o = (size_t)r * Dn + t;

    float xv = x[o];
    float mv = MX[o];
    float xn  = fmaxf(sqrtf(blockSum128(xv * xv, sm)), MINN);
    float mxn = fmaxf(sqrtf(blockSum128(mv * mv, sm)), MINN);
    float nz  = blockSum128(mv != 0.0f ? 1.0f : 0.0f, sm);

    float res = tanhf(mxn / xn * artanh_c(xn)) * mv / mxn;   // sc = 1
    if (nz == 0.0f) res = 0.0f;                              // zero_row case
    float rn = fmaxf(sqrtf(blockSum128(res * res, sm)), MINN);
    if (rn > MAXN) res *= MAXN / rn;                         // proj

    // hyperbolic bias: proj(expmap0(b_lin))
    float bv = b_lin[t];
    float bn = fmaxf(sqrtf(blockSum128(bv * bv, sm)), MINN);
    float eb = tanhf(bn) * bv / bn;
    float ebn = fmaxf(sqrtf(blockSum128(eb * eb, sm)), MINN);
    if (ebn > MAXN) eb *= MAXN / ebn;

    // mobius_add(res, eb)
    float x2 = blockSum128(res * res, sm);
    float y2 = blockSum128(eb * eb, sm);
    float xy = blockSum128(res * eb, sm);
    float num = (1.0f + 2.0f * xy + y2) * res + (1.0f - x2) * eb;
    float den = fmaxf(1.0f + 2.0f * xy + x2 * y2, MINN);
    float h = num / den;
    float hn = fmaxf(sqrtf(blockSum128(h * h, sm)), MINN);
    if (hn > MAXN) h *= MAXN / hn;                           // proj

    float hh2 = blockSum128(h * h, sm);
    H[o] = h; Hh[o] = (_Float16)h;
    if (t == 0) h2out[r] = hh2;

    // x_self = logmap(h, h): sub = mobius_add(-h, h)
    float a   = 1.0f + 2.0f * (-hh2) + hh2;
    float bc  = 1.0f - hh2;
    float dn2 = fmaxf(1.0f + 2.0f * (-hh2) + hh2 * hh2, MINN);
    float sub = (a * (-h) + bc * h) / dn2;
    float sn  = fmaxf(sqrtf(blockSum128(sub * sub, sm)), MINN);
    float lam = 2.0f / fmaxf(1.0f - hh2, MINN);
    float xs  = (2.0f / lam) * artanh_c(sn) * sub / sn;
    Xself[o] = xs; Xselfh[o] = (_Float16)xs;
}

// ---------------- pairwise attention (factored x_loc) ----------------
// x_loc(i,j) = u1*h_i + u2*h_j ; att = sigmoid(w2 . silu(u1*Pb_i + u2*Pb_j + Qa_i + b1) + b2)
// emit C2[b,i,j] = att*mask*u2 and c1[b,i] = sum_j att*mask*u1
__global__ void k_attention(const float* __restrict__ G, const float* __restrict__ h2,
                            const float* __restrict__ Pb, const float* __restrict__ Qa,
                            const float* __restrict__ att_b1, const float* __restrict__ att_w2,
                            const float* __restrict__ att_b2, const float* __restrict__ mask,
                            float* __restrict__ C2, float* __restrict__ c1)
{
    const int i  = blockIdx.x;           // node index
    const int bb = blockIdx.y;           // batch index
    const int t  = threadIdx.x;          // 0..255 (8 waves of 32)
    const int lane = t & 31, w = t >> 5;
    __shared__ float sPb[128], sQa[128], sB1[128], sW2[128], sC1[8];
    const int r = bb * Nn + i;
    if (t < 128) {
        sPb[t] = Pb[(size_t)r * Dn + t];
        sQa[t] = Qa[(size_t)r * Dn + t];
        sB1[t] = att_b1[t];
        sW2[t] = att_w2[t];
    }
    __syncthreads();

    const float hi2 = h2[r];
    const float b2v = att_b2[0];
    float c1w = 0.0f;

    for (int j = w; j < Nn; j += 8) {
        const int rj = bb * Nn + j;
        float hj2 = h2[rj];
        float g   = G[((size_t)bb * Nn + i) * Nn + j];       // <h_i, h_j>
        // mobius_add(-h_i, h_j): xy = -g, x2 = hi2, y2 = hj2
        float Ac  = 1.0f - 2.0f * g + hj2;
        float Bc  = 1.0f - hi2;
        float den = fmaxf(1.0f - 2.0f * g + hi2 * hj2, MINN);
        float s1 = -Ac / den, s2 = Bc / den;
        float sn2 = s1 * s1 * hi2 + 2.0f * s1 * s2 * g + s2 * s2 * hj2;
        float sn  = fmaxf(sqrtf(fmaxf(sn2, 0.0f)), MINN);
        float lam = 2.0f / fmaxf(1.0f - hi2, MINN);
        float gam = (2.0f / lam) * artanh_c(sn) / sn;        // logmap scale
        float u1 = gam * s1, u2 = gam * s2;                  // x_loc = u1*h_i + u2*h_j

        float partial = 0.0f;
        #pragma unroll
        for (int tt = 0; tt < 4; ++tt) {
            int k = lane * 4 + tt;
            float v = u1 * sPb[k] + u2 * Pb[(size_t)rj * Dn + k] + sQa[k] + sB1[k];
            partial += siluf(v) * sW2[k];
        }
        #pragma unroll
        for (int off = 16; off > 0; off >>= 1)
            partial += __shfl_xor(partial, off, 32);

        float att = sigmoidf(partial + b2v);
        float wt  = att * mask[((size_t)bb * Nn + i) * Nn + j];
        if (lane == 0) {
            C2[((size_t)bb * Nn + i) * Nn + j] = wt * u2;
            c1w += wt * u1;
        }
    }
    if (lane == 0) sC1[w] = c1w;
    __syncthreads();
    if (t == 0) {
        float s = 0.0f;
        #pragma unroll
        for (int q = 0; q < 8; ++q) s += sC1[q];
        c1[r] = s;
    }
}

// ---------------- build t = concat(x_self, c1*h + S2) as f16 ----------------
__global__ void k_build_t(const float* __restrict__ Xself, const float* __restrict__ H,
                          const float* __restrict__ S2, const float* __restrict__ c1,
                          _Float16* __restrict__ Th)
{
    int idx = blockIdx.x * blockDim.x + threadIdx.x;     // BN*256
    if (idx >= BN * 2 * Dn) return;
    int r = idx >> 8, k = idx & 255;
    float v;
    if (k < Dn) v = Xself[(size_t)r * Dn + k];
    else { int kk = k - Dn; v = c1[r] * H[(size_t)r * Dn + kk] + S2[(size_t)r * Dn + kk]; }
    Th[idx] = (_Float16)v;
}

__global__ void k_silu_bias(const float* __restrict__ U, const float* __restrict__ b1,
                            _Float16* __restrict__ Uh, int n)
{
    int idx = blockIdx.x * blockDim.x + threadIdx.x;
    if (idx >= n) return;
    Uh[idx] = (_Float16)siluf(U[idx] + b1[idx & (Dn - 1)]);
}

__global__ void k_add_bias_res(const float* __restrict__ V, const float* __restrict__ b2,
                               const float* __restrict__ Xself, float* __restrict__ S, int n)
{
    int idx = blockIdx.x * blockDim.x + threadIdx.x;
    if (idx >= n) return;
    S[idx] = V[idx] + b2[idx & (Dn - 1)] + Xself[idx];
}

// ---------------- final: expmap at h, proj, HypAct ----------------
__global__ void k_final(const float* __restrict__ Sup, const float* __restrict__ H,
                        float* __restrict__ out)
{
    __shared__ float sm[128];
    const int r = blockIdx.x, t = threadIdx.x;
    const size_t o = (size_t)r * Dn + t;
    float u = Sup[o], p = H[o];

    // expmap(u, p)
    float un = fmaxf(sqrtf(blockSum128(u * u, sm)), MINN);
    float p2 = blockSum128(p * p, sm);
    float lam = 2.0f / fmaxf(1.0f - p2, MINN);
    float sec = tanhf(0.5f * lam * un) * u / un;
    float y2 = blockSum128(sec * sec, sm);
    float xy = blockSum128(p * sec, sm);
    float num = (1.0f + 2.0f * xy + y2) * p + (1.0f - p2) * sec;
    float den = fmaxf(1.0f + 2.0f * xy + p2 * y2, MINN);
    float ov = num / den;
    float on = fmaxf(sqrtf(blockSum128(ov * ov, sm)), MINN);
    if (on > MAXN) ov *= MAXN / on;                          // proj

    // HypAct: xt = silu(logmap0(ov)); out = proj(expmap0(xt))
    float on2 = fmaxf(sqrtf(blockSum128(ov * ov, sm)), MINN);
    float xt  = siluf(artanh_c(on2) * ov / on2);
    float xn  = fmaxf(sqrtf(blockSum128(xt * xt, sm)), MINN);
    float res = tanhf(xn) * xt / xn;
    float rn  = fmaxf(sqrtf(blockSum128(res * res, sm)), MINN);
    if (rn > MAXN) res *= MAXN / rn;
    out[o] = res;
}

// ================= host side =================
static inline char* bump(char*& p, size_t bytes) {
    char* r = p;
    p += (bytes + 255) & ~(size_t)255;
    return r;
}

extern "C" void kernel_launch(void* const* d_in, const int* in_sizes, int n_in,
                              void* d_out, int out_size, void* d_ws, size_t ws_size,
                              hipStream_t stream)
{
    (void)in_sizes; (void)n_in; (void)out_size; (void)ws_size;
    const float* x      = (const float*)d_in[0];
    const float* mask   = (const float*)d_in[1];
    const float* W      = (const float*)d_in[2];
    const float* b_lin  = (const float*)d_in[3];
    const float* att_w1 = (const float*)d_in[4];
    const float* att_b1 = (const float*)d_in[5];
    const float* att_w2 = (const float*)d_in[6];
    const float* att_b2 = (const float*)d_in[7];
    const float* mlp_w1 = (const float*)d_in[8];
    const float* mlp_b1 = (const float*)d_in[9];
    const float* mlp_w2 = (const float*)d_in[10];
    const float* mlp_b2 = (const float*)d_in[11];
    float* out = (float*)d_out;

    char* p = (char*)d_ws;
    float* MX    = (float*)bump(p, (size_t)BN * Dn * 4);   // also reused as V later
    float* H     = (float*)bump(p, (size_t)BN * Dn * 4);
    float* h2    = (float*)bump(p, (size_t)BN * 4);
    float* Xself = (float*)bump(p, (size_t)BN * Dn * 4);
    float* Pb    = (float*)bump(p, (size_t)BN * Dn * 4);
    float* Qa    = (float*)bump(p, (size_t)BN * Dn * 4);
    float* G     = (float*)bump(p, (size_t)Bn * Nn * Nn * 4);
    float* C2    = (float*)bump(p, (size_t)Bn * Nn * Nn * 4);
    float* c1    = (float*)bump(p, (size_t)BN * 4);
    float* S2    = (float*)bump(p, (size_t)BN * Dn * 4);
    float* U     = (float*)bump(p, (size_t)BN * Dn * 4);
    float* Sup   = (float*)bump(p, (size_t)BN * Dn * 4);
    _Float16* xh      = (_Float16*)bump(p, (size_t)BN * Dn * 2);
    _Float16* Wh      = (_Float16*)bump(p, (size_t)Dn * Dn * 2);
    _Float16* Hh      = (_Float16*)bump(p, (size_t)BN * Dn * 2);
    _Float16* HhT     = (_Float16*)bump(p, (size_t)BN * Dn * 2);   // per-batch H^T (Dn x Nn)
    _Float16* Xselfh  = (_Float16*)bump(p, (size_t)BN * Dn * 2);
    _Float16* attw1h  = (_Float16*)bump(p, (size_t)Dn * 2 * Dn * 2);
    _Float16* mlpw1h  = (_Float16*)bump(p, (size_t)Dn * 2 * Dn * 2);
    _Float16* mlpw2h  = (_Float16*)bump(p, (size_t)Dn * Dn * 2);
    _Float16* C2h     = (_Float16*)bump(p, (size_t)Bn * Nn * Nn * 2);
    _Float16* Th      = (_Float16*)bump(p, (size_t)BN * 2 * Dn * 2);
    _Float16* Uh      = (_Float16*)bump(p, (size_t)BN * Dn * 2);

    auto cvt = [&](const float* s, _Float16* d, int n) {
        k_cvt_h<<<(n + 255) / 256, 256, 0, stream>>>(s, d, n);
    };

    // 0) f16 copies of GEMM operands
    cvt(x, xh, BN * Dn);
    cvt(W, Wh, Dn * Dn);
    cvt(att_w1, attw1h, Dn * 2 * Dn);
    cvt(mlp_w1, mlpw1h, Dn * 2 * Dn);
    cvt(mlp_w2, mlpw2h, Dn * Dn);

    // 1) MX = X @ W^T                         [1024 x 128] x [128 x 128]
    k_gemm_wmma<<<dim3(BN / 16, Dn / 16, 1), 32, 0, stream>>>(
        xh, 0, Dn, Wh, 0, Dn, MX, 0, Dn, Dn);

    // 2) hyperbolic linear pointwise -> H, Hh, h2, Xself
    k_hyplinear<<<BN, 128, 0, stream>>>(x, MX, b_lin, H, Hh, h2, Xself, Xselfh);

    // 3) Gram G_b = H_b H_b^T  (batched over z=4)
    k_gemm_wmma<<<dim3(Nn / 16, Nn / 16, Bn), 32, 0, stream>>>(
        Hh, (long long)Nn * Dn, Dn, Hh, (long long)Nn * Dn, Dn,
        G, (long long)Nn * Nn, Nn, Dn);

    // 4) Pb = H @ att_w1[:,D:]^T ; Qa = Xself @ att_w1[:,:D]^T
    k_gemm_wmma<<<dim3(BN / 16, Dn / 16, 1), 32, 0, stream>>>(
        Hh, 0, Dn, attw1h + Dn, 0, 2 * Dn, Pb, 0, Dn, Dn);
    k_gemm_wmma<<<dim3(BN / 16, Dn / 16, 1), 32, 0, stream>>>(
        Xselfh, 0, Dn, attw1h, 0, 2 * Dn, Qa, 0, Dn, Dn);

    // 5) pairwise attention coefficients
    k_attention<<<dim3(Nn, Bn), 256, 0, stream>>>(
        G, h2, Pb, Qa, att_b1, att_w2, att_b2, mask, C2, c1);

    // 6) S2_b = C2_b @ H_b : transpose H_b (f16, LDS-tiled), then fast-path GEMM
    cvt(C2, C2h, Bn * Nn * Nn);
    k_transpose_h<<<dim3(Nn / 32, Dn / 32, Bn), dim3(32, 8), 0, stream>>>(Hh, HhT);
    k_gemm_wmma<<<dim3(Nn / 16, Dn / 16, Bn), 32, 0, stream>>>(
        C2h, (long long)Nn * Nn, Nn, HhT, (long long)Nn * Dn, Nn,
        S2, (long long)Nn * Dn, Dn, Nn);

    // 7) node MLP: t = [x_self, c1*h + S2];  U = t @ mlp_w1^T
    {
        int n = BN * 2 * Dn;
        k_build_t<<<(n + 255) / 256, 256, 0, stream>>>(Xself, H, S2, c1, Th);
    }
    k_gemm_wmma<<<dim3(BN / 16, Dn / 16, 1), 32, 0, stream>>>(
        Th, 0, 2 * Dn, mlpw1h, 0, 2 * Dn, U, 0, Dn, 2 * Dn);
    k_silu_bias<<<(BN * Dn + 255) / 256, 256, 0, stream>>>(U, mlp_b1, Uh, BN * Dn);
    // V (reuse MX) = silu(U) @ mlp_w2^T
    k_gemm_wmma<<<dim3(BN / 16, Dn / 16, 1), 32, 0, stream>>>(
        Uh, 0, Dn, mlpw2h, 0, Dn, MX, 0, Dn, Dn);
    k_add_bias_res<<<(BN * Dn + 255) / 256, 256, 0, stream>>>(MX, mlp_b2, Xself, Sup, BN * Dn);

    // 8) expmap at h, proj, HypAct
    k_final<<<BN, 128, 0, stream>>>(Sup, H, out);
}
